// MGRFN_88347477279417
// MI455X (gfx1250) — matrix operand order repaired and checked by reading, not compiled, not run on hardware
//
#include <hip/hip_runtime.h>
#include <hip/hip_bf16.h>

// ---------------------------------------------------------------------------
// DimeNet++ interaction block for MI455X (gfx1250, wave32, WMMA).
// Weights pre-swizzled into WMMA B-fragment layout (32B/lane vector loads);
// B fragments preloaded in groups of 4 for load->wmma pipelining.
// ---------------------------------------------------------------------------

typedef __attribute__((ext_vector_type(16))) __bf16 v16bf;
typedef __attribute__((ext_vector_type(4)))  __bf16 v4bf;
typedef __attribute__((ext_vector_type(2)))  __bf16 v2bf;
typedef __attribute__((ext_vector_type(8)))  float  v8f;

constexpr int HD  = 128;   // hidden channels
constexpr int IC  = 64;    // int_emb_size
constexpr int SK  = 42;    // NS*NR
constexpr int SKP = 64;    // padded
constexpr int TK  = 294;   // NS*NS*NR
constexpr int TKP = 320;   // padded (multiple of 32)

// bf16 weight arena layout (element offsets); every weight stored fragment-swizzled
constexpr int OFF_WJI   = 0;
constexpr int OFF_WKJ   = OFF_WJI  + HD * HD;
constexpr int OFF_WDOWN = OFF_WKJ  + HD * HD;
constexpr int OFF_WUP   = OFF_WDOWN + HD * IC;
constexpr int OFF_WLIN  = OFF_WUP  + IC * HD;
constexpr int OFF_WRB1  = OFF_WLIN + HD * HD;
constexpr int OFF_WRB2  = OFF_WRB1 + HD * HD;
constexpr int OFF_WRA1  = OFF_WRB2 + HD * HD;       // 2 layers back-to-back
constexpr int OFF_WRA2  = OFF_WRA1 + 2 * HD * HD;
constexpr int OFF_WSBF  = OFF_WRA2 + 2 * HD * HD;   // padded [64][64]
constexpr int OFF_WT    = OFF_WSBF + SKP * IC;      // padded [320][64]
constexpr int WB_TOTAL  = OFF_WT   + TKP * IC;

__device__ __forceinline__ v8f vzero() {
  v8f z = {0.f, 0.f, 0.f, 0.f, 0.f, 0.f, 0.f, 0.f};
  return z;
}

__device__ __forceinline__ float silu_f(float x) {
  return x * (1.0f / (1.0f + __expf(-x)));
}

// A-fragment, 16x32 bf16, from a row-major staging buffer with row stride ld.
// Layout: lane m=lane&15; g=lane>>4; element h -> k = h + 16g (matches swizzled B)
// 16 contiguous elements -> compiler emits 2x ds_load_b128.
__device__ __forceinline__ v16bf ldA(const __bf16* base, int ld, int lane) {
  const int g = lane >> 4;
  const int m = lane & 15;
  v16bf a;
#pragma unroll
  for (int h = 0; h < 16; ++h) {
    int k = h + 16 * g;
    a[h] = base[m * ld + k];
  }
  return a;
}

// B-fragment from the pre-swizzled arena: fragment (kt,nt) of a weight with
// NT = N/16 column tiles; lane's 16 bf16 are contiguous (32B aligned).
__device__ __forceinline__ v16bf ldBfrag(const __bf16* wsw, int NT, int kt, int nt, int lane) {
  return *(const v16bf*)(wsw + ((((kt * NT) + nt) * 32 + lane) << 4));
}

#define WMMA_BF16(A, B, C) \
  __builtin_amdgcn_wmma_f32_16x16x32_bf16(false, (A), false, (B), (short)0, (C), false, false)

// 4-wide pipelined WMMA group: preload 4 B fragments, then 4 WMMAs.
// ntBase is the first column tile; acc[] indexed from ntBase.
__device__ __forceinline__ void wmma_group4(const v16bf a, const __bf16* wsw, int NT, int kt,
                                            int ntBase, v8f* acc, int lane) {
  v16bf b[4];
#pragma unroll
  for (int q = 0; q < 4; ++q) b[q] = ldBfrag(wsw, NT, kt, ntBase + q, lane);
#pragma unroll
  for (int q = 0; q < 4; ++q) acc[ntBase + q] = WMMA_BF16(a, b[q], acc[ntBase + q]);
}

// ---------------------------------------------------------------------------
// prep helpers: swizzle a row-major [K][N] f32 weight into B-fragment layout.
// dst element i: h=i&15, lane=(i>>4)&31, frag=i>>9, kt=frag/NT, nt=frag%NT,
//   k = kt*32 + h + 16*(lane>>4), n = nt*16 + (lane&15)
// ---------------------------------------------------------------------------
__device__ __forceinline__ void swzB(const float* __restrict__ src, __bf16* __restrict__ dst,
                                     int K, int N, int tid, int nth) {
  const int NT = N >> 4;
  for (int i = tid; i < K * N; i += nth) {
    int h = i & 15;
    int lane = (i >> 4) & 31;
    int frag = i >> 9;
    int kt = frag / NT, nt = frag - kt * NT;
    int k = kt * 32 + h + 16 * (lane >> 4);
    int n = nt * 16 + (lane & 15);
    dst[i] = (__bf16)src[k * N + n];
  }
}

// Same, but source is the on-the-fly product W1[K x 8] @ W2[8 x N], zero-padded to Kpad.
__device__ __forceinline__ void swzB2(const float* __restrict__ W1, const float* __restrict__ W2,
                                      __bf16* __restrict__ dst, int Kreal, int Kpad, int N,
                                      int tid, int nth) {
  const int NT = N >> 4;
  for (int i = tid; i < Kpad * N; i += nth) {
    int h = i & 15;
    int lane = (i >> 4) & 31;
    int frag = i >> 9;
    int kt = frag / NT, nt = frag - kt * NT;
    int k = kt * 32 + h + 16 * (lane >> 4);
    int n = nt * 16 + (lane & 15);
    float a = 0.f;
    if (k < Kreal) {
#pragma unroll
      for (int j = 0; j < 8; ++j) a += W1[k * 8 + j] * W2[j * N + n];
    }
    dst[i] = (__bf16)a;
  }
}

__global__ void prep_kernel(const float* __restrict__ Wji, const float* __restrict__ Wkj,
                            const float* __restrict__ Wdown, const float* __restrict__ Wup,
                            const float* __restrict__ Wlin,
                            const float* __restrict__ rbW1, const float* __restrict__ rbW2,
                            const float* __restrict__ raW1, const float* __restrict__ raW2,
                            const float* __restrict__ Wsbf1, const float* __restrict__ Wsbf2,
                            const float* __restrict__ Wt1, const float* __restrict__ Wt2,
                            const float* __restrict__ Wrbf1, const float* __restrict__ Wrbf2,
                            __bf16* __restrict__ wb, float* __restrict__ Wrbf12) {
  const int tid = blockIdx.x * blockDim.x + threadIdx.x;
  const int nth = gridDim.x * blockDim.x;
  swzB(Wji,  wb + OFF_WJI,  HD, HD, tid, nth);
  swzB(Wkj,  wb + OFF_WKJ,  HD, HD, tid, nth);
  swzB(Wlin, wb + OFF_WLIN, HD, HD, tid, nth);
  swzB(rbW1, wb + OFF_WRB1, HD, HD, tid, nth);
  swzB(rbW2, wb + OFF_WRB2, HD, HD, tid, nth);
  swzB(raW1,           wb + OFF_WRA1,           HD, HD, tid, nth);
  swzB(raW1 + HD * HD, wb + OFF_WRA1 + HD * HD, HD, HD, tid, nth);
  swzB(raW2,           wb + OFF_WRA2,           HD, HD, tid, nth);
  swzB(raW2 + HD * HD, wb + OFF_WRA2 + HD * HD, HD, HD, tid, nth);
  swzB(Wdown, wb + OFF_WDOWN, HD, IC, tid, nth);
  swzB(Wup,   wb + OFF_WUP,   IC, HD, tid, nth);
  swzB2(Wsbf1, Wsbf2, wb + OFF_WSBF, SK, SKP, IC, tid, nth);
  swzB2(Wt1,   Wt2,   wb + OFF_WT,   TK, TKP, IC, tid, nth);
  // W_rbf12 = W_rbf1 @ W_rbf2  [6][128] in f32 (used scalar)
  for (int i = tid; i < 6 * HD; i += nth) {
    int k = i >> 7, n = i & 127;
    float a = 0.f;
#pragma unroll
    for (int j = 0; j < 8; ++j) a += Wrbf1[k * 8 + j] * Wrbf2[j * HD + n];
    Wrbf12[i] = a;
  }
}

// ---------------------------------------------------------------------------
// edge-front: x_ji = silu(x1@W_ji+b), x_kj_down = silu((silu(x1@W_kj+b)*rbf)@W_down)
// 4 waves/block, each wave owns a 16-edge tile. 80 WMMAs / tile.
// ---------------------------------------------------------------------------
__global__ __launch_bounds__(128)
void edge_front_kernel(const float* __restrict__ x1, const float* __restrict__ rbf0,
                       const float* __restrict__ b_ji, const float* __restrict__ b_kj,
                       const __bf16* __restrict__ wb, const float* __restrict__ Wrbf12,
                       float* __restrict__ x_ji_out, float* __restrict__ x_kj_down) {
  __shared__ __bf16 stage[4][16][HD + 8];
  __shared__ float  rst[4][16][6];
  const int lane = threadIdx.x & 31;
  const int wave = threadIdx.x >> 5;
  const int e0 = (blockIdx.x * 4 + wave) * 16;
  const int g = lane >> 4, nl = lane & 15;
  __bf16(*st)[HD + 8] = stage[wave];

  // stage x1 tile (float4 -> v4bf, 8B LDS stores)
  for (int i = lane; i < 16 * (HD / 4); i += 32) {
    int r = i >> 5, c = (i & 31) * 4;
    const float4 v = *(const float4*)&x1[(size_t)(e0 + r) * HD + c];
    v4bf b = {(__bf16)v.x, (__bf16)v.y, (__bf16)v.z, (__bf16)v.w};
    *(v4bf*)&st[r][c] = b;
  }
  // stage rbf0 tile [16][6]
  for (int i = lane; i < 16 * 6; i += 32) {
    int r = i / 6, c = i - r * 6;
    rst[wave][r][c] = rbf0[(size_t)(e0 + r) * 6 + c];
  }
  __syncthreads();

  v8f accj[8], acck[8];
#pragma unroll
  for (int nt = 0; nt < 8; ++nt) { accj[nt] = vzero(); acck[nt] = vzero(); }

#pragma unroll
  for (int kt = 0; kt < 4; ++kt) {
    v16bf a = ldA(&st[0][0] + kt * 32, HD + 8, lane);
    wmma_group4(a, wb + OFF_WJI, 8, kt, 0, accj, lane);
    wmma_group4(a, wb + OFF_WJI, 8, kt, 4, accj, lane);
    wmma_group4(a, wb + OFF_WKJ, 8, kt, 0, acck, lane);
    wmma_group4(a, wb + OFF_WKJ, 8, kt, 4, acck, lane);
  }
  __syncthreads();

  // epilogue: write x_ji, restage (silu(x_kj) * rbf) as bf16 for down-projection
#pragma unroll
  for (int nt = 0; nt < 8; ++nt) {
#pragma unroll
    for (int v = 0; v < 8; ++v) {
      int m = v + 8 * g;
      int n = nt * 16 + nl;
      float xj = silu_f(accj[nt][v] + b_ji[n]);
      x_ji_out[(size_t)(e0 + m) * HD + n] = xj;
      float xk = silu_f(acck[nt][v] + b_kj[n]);
      float rb = 0.f;
#pragma unroll
      for (int j = 0; j < 6; ++j)
        rb += rst[wave][m][j] * Wrbf12[j * HD + n];
      st[m][n] = (__bf16)(xk * rb);
    }
  }
  __syncthreads();

  v8f accd[4];
#pragma unroll
  for (int nt = 0; nt < 4; ++nt) accd[nt] = vzero();
#pragma unroll
  for (int kt = 0; kt < 4; ++kt) {
    v16bf a = ldA(&st[0][0] + kt * 32, HD + 8, lane);
    wmma_group4(a, wb + OFF_WDOWN, 4, kt, 0, accd, lane);
  }
#pragma unroll
  for (int nt = 0; nt < 4; ++nt)
#pragma unroll
    for (int v = 0; v < 8; ++v) {
      int m = v + 8 * g;
      int n = nt * 16 + nl;
      x_kj_down[(size_t)(e0 + m) * IC + n] = silu_f(accd[nt][v]);
    }
}

// ---------------------------------------------------------------------------
// triplet: sbf_e/t_e GEMMs (pre-combined, pre-swizzled weights), gather x_kj,
// scatter-add into agg. Streams t (1.18 GB) once. 48 WMMAs / 16-triplet tile.
// ---------------------------------------------------------------------------
__global__ __launch_bounds__(128)
void triplet_kernel(const float* __restrict__ sbf, const float* __restrict__ t,
                    const int* __restrict__ idx_kj, const int* __restrict__ idx_ji,
                    const __bf16* __restrict__ wb, const float* __restrict__ x_kj_down,
                    float* __restrict__ agg, int T) {
  __shared__ __bf16 tst[4][16][TKP];   // 40 KB
  __shared__ __bf16 sst[4][16][SKP];   // 8 KB
  const int lane = threadIdx.x & 31;
  const int wave = threadIdx.x >> 5;
  const int t0 = (blockIdx.x * 4 + wave) * 16;
  const int g = lane >> 4, nl = lane & 15;

  if (t0 + 16 < T)  // gfx1250 global_prefetch_b8 for the next tile's t rows
    __builtin_prefetch(&t[(size_t)(t0 + 16) * TK], 0, 1);

  // stage t tile: rows are 1176B (8B aligned) -> float2 vector loads; TK=294 is even
  for (int i = lane; i < 16 * (TK / 2); i += 32) {
    int r = i / (TK / 2), c = (i - r * (TK / 2)) * 2;
    const float2 v = *(const float2*)&t[(size_t)(t0 + r) * TK + c];
    v2bf b = {(__bf16)v.x, (__bf16)v.y};
    *(v2bf*)&tst[wave][r][c] = b;
  }
  // zero-fill K padding 294..319
  for (int i = lane; i < 16 * ((TKP - TK) / 2); i += 32) {
    int r = i / 13, c = TK + (i - r * 13) * 2;
    v2bf z = {(__bf16)0.f, (__bf16)0.f};
    *(v2bf*)&tst[wave][r][c] = z;
  }
  // stage sbf tile: SK=42 even, rows 168B (8B aligned)
  for (int i = lane; i < 16 * (SK / 2); i += 32) {
    int r = i / (SK / 2), c = (i - r * (SK / 2)) * 2;
    const float2 v = *(const float2*)&sbf[(size_t)(t0 + r) * SK + c];
    v2bf b = {(__bf16)v.x, (__bf16)v.y};
    *(v2bf*)&sst[wave][r][c] = b;
  }
  for (int i = lane; i < 16 * ((SKP - SK) / 2); i += 32) {
    int r = i / 11, c = SK + (i - r * 11) * 2;
    v2bf z = {(__bf16)0.f, (__bf16)0.f};
    *(v2bf*)&sst[wave][r][c] = z;
  }
  __syncthreads();

  v8f acct[4], accs[4];
#pragma unroll
  for (int nt = 0; nt < 4; ++nt) { acct[nt] = vzero(); accs[nt] = vzero(); }

#pragma unroll
  for (int kt = 0; kt < TKP / 32; ++kt) {
    v16bf a = ldA(&tst[wave][0][0] + kt * 32, TKP, lane);
    wmma_group4(a, wb + OFF_WT, 4, kt, 0, acct, lane);
  }
#pragma unroll
  for (int kt = 0; kt < SKP / 32; ++kt) {
    v16bf a = ldA(&sst[wave][0][0] + kt * 32, SKP, lane);
    wmma_group4(a, wb + OFF_WSBF, 4, kt, 0, accs, lane);
  }

  // msg = x_kj_down[idx_kj] * sbf_e * t_e ; agg[idx_ji] += msg
#pragma unroll
  for (int v = 0; v < 8; ++v) {
    int m = v + 8 * g;
    int kj = idx_kj[t0 + m];
    int ji = idx_ji[t0 + m];
#pragma unroll
    for (int nt = 0; nt < 4; ++nt) {
      int n = nt * 16 + nl;
      float val = acct[nt][v] * accs[nt][v] * x_kj_down[(size_t)kj * IC + n];
      unsafeAtomicAdd(&agg[(size_t)ji * IC + n], val);  // global_atomic_add_f32
    }
  }
}

// ---------------------------------------------------------------------------
// edge-back: e1 = x_ji + silu(agg@W_up); residual stacks; e2 = (rbf0@W_rbf)*e1
// 240 WMMAs / 16-edge tile.
// ---------------------------------------------------------------------------
__global__ __launch_bounds__(128)
void edge_back_kernel(const float* __restrict__ x1, const float* __restrict__ rbf0,
                      const float* __restrict__ W_rbf, const float* __restrict__ agg,
                      const __bf16* __restrict__ wb,
                      const float* __restrict__ b_lin,
                      const float* __restrict__ resb_b1, const float* __restrict__ resb_b2,
                      const float* __restrict__ resa_b1, const float* __restrict__ resa_b2,
                      float* __restrict__ e1_io, float* __restrict__ e2_out) {
  __shared__ __bf16 stage[4][16][HD + 8];
  __shared__ float  rst[4][16][6];
  const int lane = threadIdx.x & 31;
  const int wave = threadIdx.x >> 5;
  const int e0 = (blockIdx.x * 4 + wave) * 16;
  const int g = lane >> 4, nl = lane & 15;
  __bf16(*st)[HD + 8] = stage[wave];

  // stage agg tile (float4 -> v4bf) for up-projection
  for (int i = lane; i < 16 * (IC / 4); i += 32) {
    int r = i >> 4, c = (i & 15) * 4;
    const float4 v = *(const float4*)&agg[(size_t)(e0 + r) * IC + c];
    v4bf b = {(__bf16)v.x, (__bf16)v.y, (__bf16)v.z, (__bf16)v.w};
    *(v4bf*)&st[r][c] = b;
  }
  for (int i = lane; i < 16 * 6; i += 32) {
    int r = i / 6, c = i - r * 6;
    rst[wave][r][c] = rbf0[(size_t)(e0 + r) * 6 + c];
  }
  __syncthreads();

  v8f acc[8];
#pragma unroll
  for (int nt = 0; nt < 8; ++nt) acc[nt] = vzero();
#pragma unroll
  for (int kt = 0; kt < 2; ++kt) {
    v16bf a = ldA(&st[0][0] + kt * 32, HD + 8, lane);
    wmma_group4(a, wb + OFF_WUP, 8, kt, 0, acc, lane);
    wmma_group4(a, wb + OFF_WUP, 8, kt, 4, acc, lane);
  }

  float e1r[8][8];
#pragma unroll
  for (int nt = 0; nt < 8; ++nt)
#pragma unroll
    for (int v = 0; v < 8; ++v) {
      int m = v + 8 * g, n = nt * 16 + nl;
      // e1_io currently holds x_ji (written by edge_front)
      e1r[nt][v] = e1_io[(size_t)(e0 + m) * HD + n] + silu_f(acc[nt][v]);
    }

  auto restage = [&](const float vals[8][8]) {
    __syncthreads();
#pragma unroll
    for (int nt = 0; nt < 8; ++nt)
#pragma unroll
      for (int v = 0; v < 8; ++v)
        st[v + 8 * g][nt * 16 + nl] = (__bf16)vals[nt][v];
    __syncthreads();
  };
  auto gemm128 = [&](const __bf16* W, v8f out[8]) {
#pragma unroll
    for (int nt = 0; nt < 8; ++nt) out[nt] = vzero();
#pragma unroll
    for (int kt = 0; kt < 4; ++kt) {
      v16bf a = ldA(&st[0][0] + kt * 32, HD + 8, lane);
      wmma_group4(a, W, 8, kt, 0, out, lane);
      wmma_group4(a, W, 8, kt, 4, out, lane);
    }
  };
  auto reslayer = [&](const __bf16* W1, const float* b1, const __bf16* W2, const float* b2) {
    restage(e1r);
    v8f a1[8];
    gemm128(W1, a1);
    float h[8][8];
#pragma unroll
    for (int nt = 0; nt < 8; ++nt)
#pragma unroll
      for (int v = 0; v < 8; ++v)
        h[nt][v] = silu_f(a1[nt][v] + b1[nt * 16 + nl]);
    restage(h);
    v8f a2[8];
    gemm128(W2, a2);
#pragma unroll
    for (int nt = 0; nt < 8; ++nt)
#pragma unroll
      for (int v = 0; v < 8; ++v)
        e1r[nt][v] += silu_f(a2[nt][v] + b2[nt * 16 + nl]);
  };

  // residual before skip (NBS = 1)
  reslayer(wb + OFF_WRB1, resb_b1, wb + OFF_WRB2, resb_b2);

  // e1 = silu(e1@W_lin + b_lin) + x1
  restage(e1r);
  v8f al[8];
  gemm128(wb + OFF_WLIN, al);
#pragma unroll
  for (int nt = 0; nt < 8; ++nt)
#pragma unroll
    for (int v = 0; v < 8; ++v) {
      int m = v + 8 * g, n = nt * 16 + nl;
      e1r[nt][v] = silu_f(al[nt][v] + b_lin[n]) + x1[(size_t)(e0 + m) * HD + n];
    }

  // residuals after skip (NAS = 2)
  reslayer(wb + OFF_WRA1, resa_b1, wb + OFF_WRA2, resa_b2);
  reslayer(wb + OFF_WRA1 + HD * HD, resa_b1 + HD, wb + OFF_WRA2 + HD * HD, resa_b2 + HD);

  // outputs: e1 and e2 = (rbf0 @ W_rbf) * e1
#pragma unroll
  for (int nt = 0; nt < 8; ++nt)
#pragma unroll
    for (int v = 0; v < 8; ++v) {
      int m = v + 8 * g, n = nt * 16 + nl;
      float ev = e1r[nt][v];
      e1_io[(size_t)(e0 + m) * HD + n] = ev;
      float rb = 0.f;
#pragma unroll
      for (int j = 0; j < 6; ++j)
        rb += rst[wave][m][j] * W_rbf[j * HD + n];
      e2_out[(size_t)(e0 + m) * HD + n] = rb * ev;
    }
}

// ---------------------------------------------------------------------------
extern "C" void kernel_launch(void* const* d_in, const int* in_sizes, int n_in,
                              void* d_out, int out_size, void* d_ws, size_t ws_size,
                              hipStream_t stream) {
  const float* x1      = (const float*)d_in[0];
  const float* rbf0    = (const float*)d_in[1];
  const float* sbf     = (const float*)d_in[2];
  const float* t       = (const float*)d_in[3];
  const int*   idx_kj  = (const int*)d_in[4];
  const int*   idx_ji  = (const int*)d_in[5];
  const float* W_rbf1  = (const float*)d_in[6];
  const float* W_rbf2  = (const float*)d_in[7];
  const float* W_sbf1  = (const float*)d_in[8];
  const float* W_sbf2  = (const float*)d_in[9];
  const float* W_t1    = (const float*)d_in[10];
  const float* W_t2    = (const float*)d_in[11];
  const float* W_rbf   = (const float*)d_in[12];
  const float* W_kj    = (const float*)d_in[13];
  const float* b_kj    = (const float*)d_in[14];
  const float* W_ji    = (const float*)d_in[15];
  const float* b_ji    = (const float*)d_in[16];
  const float* W_down  = (const float*)d_in[17];
  const float* W_up    = (const float*)d_in[18];
  const float* W_lin   = (const float*)d_in[19];
  const float* b_lin   = (const float*)d_in[20];
  const float* resb_W1 = (const float*)d_in[21];
  const float* resb_b1 = (const float*)d_in[22];
  const float* resb_W2 = (const float*)d_in[23];
  const float* resb_b2 = (const float*)d_in[24];
  const float* resa_W1 = (const float*)d_in[25];
  const float* resa_b1 = (const float*)d_in[26];
  const float* resa_W2 = (const float*)d_in[27];
  const float* resa_b2 = (const float*)d_in[28];

  const int E = in_sizes[0] / HD;   // 200000 (divisible by 64)
  const int T = in_sizes[4];        // 1000000 (divisible by 64)

  // workspace carve-up
  float*  x_kj_down = (float*)d_ws;                      // E*64 f32
  float*  agg       = x_kj_down + (size_t)E * IC;        // E*64 f32
  __bf16* wb        = (__bf16*)(agg + (size_t)E * IC);   // swizzled bf16 weight arena
  float*  Wrbf12    = (float*)(wb + WB_TOTAL);           // [6][128] f32

  float* e1 = (float*)d_out;            // x_ji staging, then final e1
  float* e2 = e1 + (size_t)E * HD;      // final e2

  (void)hipMemsetAsync(agg, 0, (size_t)E * IC * sizeof(float), stream);
  prep_kernel<<<96, 256, 0, stream>>>(W_ji, W_kj, W_down, W_up, W_lin,
                                      resb_W1, resb_W2, resa_W1, resa_W2,
                                      W_sbf1, W_sbf2, W_t1, W_t2,
                                      W_rbf1, W_rbf2, wb, Wrbf12);
  edge_front_kernel<<<E / 64, 128, 0, stream>>>(x1, rbf0, b_ji, b_kj, wb, Wrbf12,
                                                e1, x_kj_down);
  triplet_kernel<<<T / 64, 128, 0, stream>>>(sbf, t, idx_kj, idx_ji, wb,
                                             x_kj_down, agg, T);
  edge_back_kernel<<<E / 64, 128, 0, stream>>>(x1, rbf0, W_rbf, agg, wb,
                                               b_lin, resb_b1, resb_b2,
                                               resa_b1, resa_b2, e1, e2);
}
